// MemKDMLayer_74285754352159
// MI455X (gfx1250) — matrix operand order repaired and checked by reading, not compiled, not run on hardware
//
#include <hip/hip_runtime.h>
#include <hip/hip_bf16.h>

typedef __attribute__((ext_vector_type(16))) __bf16 v16bf;
typedef __attribute__((ext_vector_type(8)))  float  v8f;
typedef __attribute__((ext_vector_type(4)))  unsigned int u32x4;
typedef float f32x4u __attribute__((ext_vector_type(4), aligned(4)));  // dword-aligned vec4 load

#define BS   64
#define NCI  64      // i (rows of in_v)
#define NC   1024    // j (neighbors)
#define DX   512     // K
#define DY   128
#define JT   128     // j-tile per workgroup
#define KC   32      // K chunk per WMMA step
#define NKC  (DX / KC)
#define ASTR 40      // bf16 elems per LDS row (32 + 8 pad -> 80B stride, 16B aligned)

// pack two f32 -> bf16 pair: round-to-nearest (+0x8000) then v_perm_b32 selects
// the two high halves in a single instruction.
__device__ __forceinline__ unsigned int pk2(float a, float b) {
  unsigned int ua = __float_as_uint(a) + 0x8000u;
  unsigned int ub = __float_as_uint(b) + 0x8000u;
  return __builtin_amdgcn_perm(ub, ua, 0x07060302u);   // {ub[31:16], ua[31:16]}
}

__global__ void __launch_bounds__(256) zero_rowsum(float* rs) {
  int i = blockIdx.x * 256 + threadIdx.x;
  if (i < BS * NCI) rs[i] = 0.f;
}

// One WG = (batch b, j-tile of 128). M=64, N=128, K=512 bf16 WMMA GEMM fused
// with the RBF epilogue. Double-buffered LDS staging (f32 -> bf16 on the fly),
// one barrier per K-chunk. Writes W[b,i,j]; atomically accumulates rowsum[b,i].
__global__ void __launch_bounds__(256) rbf_gemm(const float* __restrict__ rho,
                                                const float* __restrict__ nbr,
                                                float* __restrict__ rowsum,
                                                float* __restrict__ Wout) {
  __shared__ __align__(16) unsigned short Abuf[2][NCI * ASTR];
  __shared__ __align__(16) unsigned short Bbuf[2][JT  * ASTR];
  __shared__ float sA2[NCI];
  __shared__ float sB2[JT];

  const int tid = threadIdx.x;
  const int b   = blockIdx.x >> 3;
  const int jt  = blockIdx.x & 7;

  if (tid < NCI) sA2[tid] = 0.f;
  if (tid < JT)  sB2[tid] = 0.f;

  // staging assignment: A 64x32 -> 8 f32/thread, B 128x32 -> 16 f32/thread
  const int rowA  = tid & 63;
  const int koffA = (tid >> 6) << 3;            // 0,8,16,24
  const int rowB  = tid & 127;
  const int koffB = (tid >> 7) << 4;            // 0,16
  const float* Arow = rho + (size_t)(b * NCI + rowA) * (DX + 1) + 1 + koffA;
  const float* Brow = nbr + (size_t)(b * NC + jt * JT + rowB) * DX + koffB;

  // wave tiling: wave&3 -> M-tile (16 rows), wave>>2 -> N half (64 cols)
  const int wave  = tid >> 5;
  const int lane  = tid & 31;
  const int hi    = lane >> 4;
  const int ln    = lane & 15;
  const int Mbase = (wave & 3) * 16;
  const int n2    = (wave >> 2) * 64;

  // fragment byte offsets within a buffer
  const int aOff = (Mbase + ln) * (ASTR * 2) + hi * 16;  // A: lo K0-7/16-23, hi K8-15/24-31
  int bOff[4];
  #pragma unroll
  for (int t = 0; t < 4; ++t)                            // B: lo K0-7/16-23, hi K8-15/24-31
    bOff[t] = (n2 + t * 16 + ln) * (ASTR * 2) + hi * 16;

  v8f acc[4] = {};
  float a2p = 0.f, b2p = 0.f;

  f32x4u av[2];
  f32x4u bv[4];
  auto gload = [&](int kcc) {                  // global load + norm accumulation
    const int kb = kcc * KC;
    av[0] = *(const f32x4u*)(Arow + kb);       // dword-aligned b128 (row base has +1 col)
    av[1] = *(const f32x4u*)(Arow + kb + 4);
    #pragma unroll
    for (int q = 0; q < 4; ++q) bv[q] = *(const f32x4u*)(Brow + kb + q * 4);
    #pragma unroll
    for (int p = 0; p < 2; ++p)
      a2p += av[p].x * av[p].x + av[p].y * av[p].y +
             av[p].z * av[p].z + av[p].w * av[p].w;
    #pragma unroll
    for (int q = 0; q < 4; ++q)
      b2p += bv[q].x * bv[q].x + bv[q].y * bv[q].y +
             bv[q].z * bv[q].z + bv[q].w * bv[q].w;
  };
  auto lstore = [&](int buf) {                 // pack bf16 + LDS store
    u32x4 pa;
    pa.x = pk2(av[0].x, av[0].y); pa.y = pk2(av[0].z, av[0].w);
    pa.z = pk2(av[1].x, av[1].y); pa.w = pk2(av[1].z, av[1].w);
    *(u32x4*)((char*)Abuf[buf] + rowA * (ASTR * 2) + koffA * 2) = pa;
    u32x4 pb0, pb1;
    pb0.x = pk2(bv[0].x, bv[0].y); pb0.y = pk2(bv[0].z, bv[0].w);
    pb0.z = pk2(bv[1].x, bv[1].y); pb0.w = pk2(bv[1].z, bv[1].w);
    pb1.x = pk2(bv[2].x, bv[2].y); pb1.y = pk2(bv[2].z, bv[2].w);
    pb1.z = pk2(bv[3].x, bv[3].y); pb1.w = pk2(bv[3].z, bv[3].w);
    char* bst = (char*)Bbuf[buf] + rowB * (ASTR * 2) + koffB * 2;
    *(u32x4*)(bst)      = pb0;
    *(u32x4*)(bst + 16) = pb1;
  };

  gload(0);
  lstore(0);
  __syncthreads();

  for (int kc = 0; kc < NKC; ++kc) {
    const int cur = kc & 1;
    if (kc + 1 < NKC) gload(kc + 1);           // overlap next chunk's HBM loads

    union Frag { v16bf v; u32x4 q[2]; };
    Frag fa, fb[4];
    const char* ab = (const char*)Abuf[cur] + aOff;
    fa.q[0] = *(const u32x4*)(ab);
    fa.q[1] = *(const u32x4*)(ab + 32);
    #pragma unroll
    for (int t = 0; t < 4; ++t) {
      const char* bb = (const char*)Bbuf[cur] + bOff[t];
      fb[t].q[0] = *(const u32x4*)(bb);
      fb[t].q[1] = *(const u32x4*)(bb + 32);
    }
    #pragma unroll
    for (int t = 0; t < 4; ++t)
      acc[t] = __builtin_amdgcn_wmma_f32_16x16x32_bf16(
          false, fa.v, false, fb[t].v, (short)0, acc[t], false, false);

    if (kc + 1 < NKC) lstore(1 - cur);         // fill the other buffer
    __syncthreads();
  }

  // reduce squared norms (f32, from original data)
  atomicAdd(&sA2[rowA], a2p);
  atomicAdd(&sB2[rowB], b2p);
  __syncthreads();

  // epilogue: W = max(exp(-(a2+b2-2ab)), eps); K^2 with gamma=0.5 == exp(-d2)
  float* Wb = Wout + ((size_t)(b * NCI) << 10) + jt * JT;
  #pragma unroll
  for (int r = 0; r < 8; ++r) {
    const int i  = Mbase + r + hi * 8;   // C/D layout: lanes16-31 hold M+8
    const float a2 = sA2[i];
    float part = 0.f;
    #pragma unroll
    for (int t = 0; t < 4; ++t) {
      const int jl = n2 + t * 16 + ln;
      float d2 = a2 + sB2[jl] - 2.f * acc[t][r];
      d2 = fmaxf(d2, 0.f);
      float w = fmaxf(__expf(-d2), 1e-12f);
      Wb[((size_t)i << 10) + jl] = w;
      part += w;
    }
    part += __shfl_xor(part, 8, 16);
    part += __shfl_xor(part, 4, 16);
    part += __shfl_xor(part, 2, 16);
    part += __shfl_xor(part, 1, 16);
    if (ln == 0) atomicAdd(&rowsum[b * NCI + i], part);
  }
}

// out[b,j,0] = sum_i (in_w[b,i]/rowsum[b,i]) * W[b,i,j]
__global__ void __launch_bounds__(256) outw_kernel(const float* __restrict__ rho,
                                                   const float* __restrict__ rowsum,
                                                   const float* __restrict__ W,
                                                   float* __restrict__ out) {
  __shared__ float coef[NCI];
  const int b     = blockIdx.x >> 2;
  const int jbase = (blockIdx.x & 3) * 256;
  const int t     = threadIdx.x;
  if (t < NCI)
    coef[t] = rho[(size_t)(b * NCI + t) * (DX + 1)] / rowsum[b * NCI + t];
  __syncthreads();
  const int j = jbase + t;
  const float* Wb = W + ((size_t)(b * NCI) << 10) + j;
  float acc = 0.f;
  #pragma unroll 8
  for (int i = 0; i < NCI; ++i) acc += coef[i] * Wb[(size_t)i << 10];
  out[(size_t)(b * NC + j) * (DY + 1)] = acc;
}

__global__ void __launch_bounds__(256) label_copy(const float* __restrict__ labels,
                                                  float* __restrict__ out) {
  size_t idx = (size_t)blockIdx.x * 256 + threadIdx.x;   // grid sized exactly
  size_t bj  = idx >> 7;
  int    k   = (int)(idx & 127);
  out[bj * (DY + 1) + 1 + k] = labels[idx];
}

extern "C" void kernel_launch(void* const* d_in, const int* in_sizes, int n_in,
                              void* d_out, int out_size, void* d_ws, size_t ws_size,
                              hipStream_t stream) {
  (void)in_sizes; (void)n_in; (void)out_size; (void)ws_size;
  const float* rho    = (const float*)d_in[0];
  const float* nbr    = (const float*)d_in[1];
  const float* labels = (const float*)d_in[2];
  float* out    = (float*)d_out;
  float* rowsum = (float*)d_ws;                 // 64*64 f32
  float* W      = rowsum + BS * NCI;            // 64*64*1024 f32 (~16.8 MB)

  zero_rowsum<<<(BS * NCI + 255) / 256, 256, 0, stream>>>(rowsum);
  rbf_gemm  <<<BS * (NC / JT),       256, 0, stream>>>(rho, nbr, rowsum, W);
  outw_kernel<<<BS * (NC / 256),     256, 0, stream>>>(rho, rowsum, W, out);
  label_copy<<<(BS * NC * DY) / 256, 256, 0, stream>>>(labels, out);
}
